// Mamba3_48026324304074
// MI455X (gfx1250) — matrix operand order repaired and checked by reading, not compiled, not run on hardware
//
#include <hip/hip_runtime.h>

// Shapes fixed by the reference: B=2, S=2048, R=2, Hqk=8, H=16, dstate=128, chunk=64
#define BB   2
#define HB   16
#define RB   2
#define HQK  8
#define DSZ  128
#define CK   64
#define LDP  136   // padded LDS leading dim (halves): rows 272B -> 16B aligned

typedef __attribute__((ext_vector_type(16))) _Float16 v16h;
typedef __attribute__((ext_vector_type(8)))  _Float16 v8h;
typedef __attribute__((ext_vector_type(8)))  float    v8f;
typedef __attribute__((ext_vector_type(4)))  float    v4f;

// ---- CDNA5 WMMA (wave32) ------------------------------------------------
// A (16x32 f16): lane L row m=L%16; this code's k order: halves = run [k0+off, +16),
//                off = 16 for lanes>=16.
// B (32x16 f16): lane L col n=L%16; same k order as A: halves = run [k0+off, +16), off = 16 for lanes>=16.
// C/D (16x16 f32): vgpr g, lane L -> m = g + 8*(L>=16), n = L%16.

__device__ __forceinline__ v8f wmma_f16(v16h a, v16h b, v8f c) {
  return __builtin_amdgcn_wmma_f32_16x16x32_f16(false, a, false, b, (short)0, c,
                                                false, false);
}

__device__ __forceinline__ v16h cat16(v8h lo, v8h hi) {
  v16h r;
#pragma unroll
  for (int i = 0; i < 8; ++i) { r[i] = lo[i]; r[i + 8] = hi[i]; }
  return r;
}

// A operand from row-major f16 matrix (rows 16B-aligned): two b128 loads
__device__ __forceinline__ v16h load_a16(const _Float16* M, int ld, int row0, int k0, int lane) {
  const _Float16* p = M + (size_t)(row0 + (lane & 15)) * ld + k0 + ((lane >= 16) ? 16 : 0);
  return cat16(*(const v8h*)p, *(const v8h*)(p + 8));
}
// B operand from n-major f16 matrix (contraction dim contiguous): two b128 loads
__device__ __forceinline__ v16h load_b16(const _Float16* M, int ld, int col0, int k0, int lane) {
  const _Float16* p = M + (size_t)(col0 + (lane & 15)) * ld + k0 + ((lane >= 16) ? 16 : 0);
  return cat16(*(const v8h*)p, *(const v8h*)(p + 8));
}
// B operand from n-major f32 matrix: four b128 loads + packed converts
__device__ __forceinline__ v16h load_b16_f32(const float* M, int ld, int col0, int k0, int lane) {
  const float* p = M + (size_t)(col0 + (lane & 15)) * ld + k0 + ((lane >= 16) ? 16 : 0);
  v16h b;
#pragma unroll
  for (int i = 0; i < 16; i += 4) {
    v4f x = *(const v4f*)(p + i);
#pragma unroll
    for (int j = 0; j < 4; ++j) b[i + j] = (_Float16)x[j];
  }
  return b;
}

// ---------------------------------------------------------------------------
// K0: per-(b,h,chunk) scans: dA cumsum, suffix exp, decay; k-factor & shifted_gamma
__global__ void k_pre(const float* __restrict__ da, const float* __restrict__ dt,
                      const float* __restrict__ trap,
                      float* __restrict__ a_cs, float* __restrict__ e_rev,
                      float* __restrict__ factor, float* __restrict__ sg_out,
                      float* __restrict__ decay, int S, int nch) {
  int blk = blockIdx.x;
  int c  = blk % nch;
  int bh = blk / nch;
  int t  = threadIdx.x;  // 0..63
  __shared__ float lda[CK];
  int s    = c * CK + t;
  int gidx = bh * S + s;
  lda[t] = da[gidx];
  __syncthreads();
  float fwd = 0.f, suf = 0.f;
  for (int i = 0; i <= t; ++i) fwd += lda[i];
  for (int i = t + 1; i < CK; ++i) suf += lda[i];
  float acs = fminf(fwd, 0.f);
  a_cs[gidx]  = acs;
  e_rev[gidx] = __expf(fminf(suf, 0.f));
  if (t == CK - 1) decay[blk] = __expf(acs);

  float dt0  = dt[gidx];
  float sig0 = 1.f / (1.f + __expf(-trap[gidx]));
  float g0   = dt0 * sig0;
  float shg  = 0.f;
  if (s + 1 < S) {
    float dt1  = dt[bh * S + s + 1];
    float sig1 = 1.f / (1.f + __expf(-trap[bh * S + s + 1]));
    shg = dt1 * (1.f - sig1);
  }
  factor[gidx] = g0 + shg;
  sg_out[gidx] = shg;
}

// ---------------------------------------------------------------------------
// K1: bias + pre-rotary qk dots + rotary + scalings -> f16 operand buffers.
//   Qh, Kh  : cr-major (CR x D)       -- A operand / B(=K^T) operand
//   Krt     : d-major  (D x CR), = (K*factor*e_rev)^T   -- KV A operand
//   Vt      : d-major  (D x CR), = (v*mimo_v)^T         -- KV/intra B operand
__global__ void k_build(const float* __restrict__ q, const float* __restrict__ k,
                        const float* __restrict__ v, const float* __restrict__ qb,
                        const float* __restrict__ kb, const float* __restrict__ mv,
                        const float* __restrict__ ang, const float* __restrict__ factor,
                        const float* __restrict__ sg, const float* __restrict__ e_rev,
                        _Float16* __restrict__ Qh, _Float16* __restrict__ Kh,
                        _Float16* __restrict__ Krt, _Float16* __restrict__ Vt,
                        float* __restrict__ qkd, int S, int nch) {
  int blk = blockIdx.x;
  int c  = blk % nch;
  int bh = blk / nch;
  int h  = bh % HB;
  int b  = bh / HB;
  int d  = threadIdx.x;       // 0..127
  int lane = d & 31, wv = d >> 5;
  int hq = h >> 1;
  size_t base = (size_t)blk * (DSZ * DSZ);

  __shared__ float qsh[RB][DSZ];
  __shared__ float ksh[RB][DSZ];
  __shared__ float wred[4][4];
  __shared__ _Float16 Tk[DSZ][72];  // 64 cr cols + pad (144B rows, 16B aligned)
  __shared__ _Float16 Tv[DSZ][72];

  for (int half = 0; half < 2; ++half) {
    for (int ii = 0; ii < 32; ++ii) {
      int ci = half * 32 + ii;
      int s  = c * CK + ci;
      float fac  = factor[bh * S + s];
      float erev = e_rev[bh * S + s];
      float sgv  = sg[bh * S + s];

      float qv0[RB], kv0[RB];
#pragma unroll
      for (int r = 0; r < RB; ++r) {
        size_t qi = (((size_t)(b * S + s) * RB + r) * HQK + hq) * DSZ + d;
        qv0[r] = q[qi] + qb[(h * RB + r) * DSZ + d];
        kv0[r] = k[qi] + kb[(h * RB + r) * DSZ + d];
        qsh[r][d] = qv0[r];
        ksh[r][d] = kv0[r];
      }
      // 2x2 pre-rotary dots via wave32 shuffle reduction
      float p00 = qv0[0] * kv0[0], p01 = qv0[0] * kv0[1];
      float p10 = qv0[1] * kv0[0], p11 = qv0[1] * kv0[1];
#pragma unroll
      for (int off = 16; off > 0; off >>= 1) {
        p00 += __shfl_down(p00, off, 32);
        p01 += __shfl_down(p01, off, 32);
        p10 += __shfl_down(p10, off, 32);
        p11 += __shfl_down(p11, off, 32);
      }
      if (lane == 0) {
        wred[wv][0] = p00; wred[wv][1] = p01; wred[wv][2] = p10; wred[wv][3] = p11;
      }
      __syncthreads();
      if (d == 0) {
        float s4[4] = {0.f, 0.f, 0.f, 0.f};
#pragma unroll
        for (int w = 0; w < 4; ++w)
#pragma unroll
          for (int e = 0; e < 4; ++e) s4[e] += wred[w][e];
        size_t qb4 = ((size_t)bh * S + s) * 4;
#pragma unroll
        for (int e = 0; e < 4; ++e) qkd[qb4 + e] = s4[e] * sgv;
      }
      // rotary-quarter + scalings
      int grp = d >> 5, j = d & 31;
      float aang = ang[((size_t)(b * S + s) * HB + h) * 32 + j];
      float cs = __cosf(aang), sn = __sinf(aang);
      float vraw = v[((size_t)(b * S + s) * HB + h) * DSZ + d];
#pragma unroll
      for (int r = 0; r < RB; ++r) {
        float qr = (grp == 0) ? qsh[r][j] * cs - qsh[r][64 + j] * sn
                 : (grp == 2) ? qsh[r][j] * sn + qsh[r][64 + j] * cs
                              : qsh[r][d];
        float kr = ((grp == 0) ? ksh[r][j] * cs - ksh[r][64 + j] * sn
                  : (grp == 2) ? ksh[r][j] * sn + ksh[r][64 + j] * cs
                               : ksh[r][d]) * fac;
        size_t row = (size_t)(ci * RB + r) * DSZ + d;
        Qh[base + row] = (_Float16)qr;
        Kh[base + row] = (_Float16)kr;
        Tk[d][ii * 2 + r] = (_Float16)(kr * erev);
        Tv[d][ii * 2 + r] = (_Float16)(vraw * mv[(h * RB + r) * DSZ + d]);
      }
      __syncthreads();
    }
    // coalesced flush of transposed 128x64 tiles (8x b128 stores per thread)
    {
      _Float16* krow = Krt + base + (size_t)d * DSZ + half * 64;
      _Float16* vrow = Vt  + base + (size_t)d * DSZ + half * 64;
#pragma unroll
      for (int x = 0; x < 64; x += 8) {
        *(v8h*)(krow + x) = *(const v8h*)&Tk[d][x];
        *(v8h*)(vrow + x) = *(const v8h*)&Tv[d][x];
      }
    }
    __syncthreads();
  }
}

// ---------------------------------------------------------------------------
// K2: per-chunk KV^T:  out[p][d] = sum_cr Krt[d][cr] * Vt[p][cr]    [WMMA]
// grid = B*H*n, 256 threads (wave w -> d rows [16w,16w+16))
__global__ void k_kv(const _Float16* __restrict__ Krt, const _Float16* __restrict__ Vt,
                     float* __restrict__ kvbuf, int nch) {
  int blk  = blockIdx.x;
  int lane = threadIdx.x & 31;
  int wave = threadIdx.x >> 5;
  size_t mbase = (size_t)blk * (DSZ * DSZ);
  const _Float16* Ka = Krt + mbase;
  const _Float16* Vb = Vt + mbase;
  int row0 = wave * 16;

  // hoist all 4 A operands (overlapped b128 loads)
  v16h aops[4];
#pragma unroll
  for (int kt = 0; kt < 4; ++kt) aops[kt] = load_a16(Ka, DSZ, row0, kt * 32, lane);

  v8f acc[8];
#pragma unroll
  for (int ct = 0; ct < 8; ++ct)
#pragma unroll
    for (int e = 0; e < 8; ++e) acc[ct][e] = 0.f;

#pragma unroll
  for (int kt = 0; kt < 4; ++kt) {
    int k0 = kt * 32;
    // batch the 8 B-operand loads so the clause keeps many b128 in flight
    v16h bops[8];
#pragma unroll
    for (int ct = 0; ct < 8; ++ct) bops[ct] = load_b16(Vb, DSZ, ct * 16, k0, lane);
#pragma unroll
    for (int ct = 0; ct < 8; ++ct) acc[ct] = wmma_f16(aops[kt], bops[ct], acc[ct]);
  }
  // store transposed (p-major) with b128 stores
  float* outp = kvbuf + mbase;
  int off8 = (lane >= 16) ? 8 : 0;
#pragma unroll
  for (int ct = 0; ct < 8; ++ct) {
    int nn = ct * 16 + (lane & 15);
    float* o = outp + (size_t)nn * DSZ + row0 + off8;
    v4f lo = {acc[ct][0], acc[ct][1], acc[ct][2], acc[ct][3]};
    v4f hi = {acc[ct][4], acc[ct][5], acc[ct][6], acc[ct][7]};
    *(v4f*)o = lo;
    *(v4f*)(o + 4) = hi;
  }
}

// ---------------------------------------------------------------------------
// K3: serial inter-chunk scan, in place: kv[c] <- pre-state for chunk c
// (latency critical: prefetch next chunk's lines while recurrence runs)
__global__ void k_scan(float* __restrict__ kvbuf, const float* __restrict__ decay, int nch) {
  int bh  = blockIdx.x;
  int tid = threadIdx.x;
  float st[64];
#pragma unroll
  for (int i = 0; i < 64; ++i) st[i] = 0.f;
  size_t base = (size_t)bh * nch * (DSZ * DSZ);
  for (int c = 0; c < nch; ++c) {
    float* p = kvbuf + base + (size_t)c * (DSZ * DSZ) + (size_t)tid * 64;
    if (c + 1 < nch) {                     // gfx1250 global_prefetch_b8
      __builtin_prefetch(p + (DSZ * DSZ), 0, 0);
      __builtin_prefetch(p + (DSZ * DSZ) + 32, 0, 0);
    }
    float dc = decay[bh * nch + c];
#pragma unroll
    for (int i = 0; i < 64; ++i) {
      float tmp = p[i];
      p[i]  = st[i];
      st[i] = dc * st[i] + tmp;
    }
  }
}

// ---------------------------------------------------------------------------
// K4: per-chunk output:  acc = diag(e^a)(Q @ states) + (mask.(Q@K^T)) @ V,
//     fused epilogue (qk correction, D*v, SiLU gate, mimo_o R-reduction)
__global__ void k_out(const _Float16* __restrict__ Qh, const _Float16* __restrict__ Kh,
                      const _Float16* __restrict__ Vt, const float* __restrict__ stT,
                      const float* __restrict__ a_cs, const float* __restrict__ qkd,
                      const float* __restrict__ v, const float* __restrict__ z,
                      const float* __restrict__ mv, const float* __restrict__ mz,
                      const float* __restrict__ mo, const float* __restrict__ Dp,
                      float* __restrict__ out, int S, int nch) {
  int blk  = blockIdx.x;
  int c    = blk % nch;
  int bh   = blk / nch;
  int h    = bh % HB;
  int b    = bh / HB;
  int lane = threadIdx.x & 31;
  int wave = threadIdx.x >> 5;
  size_t mbase = (size_t)blk * (DSZ * DSZ);
  const _Float16* Q = Qh + mbase;
  const _Float16* K = Kh + mbase;
  const _Float16* V = Vt + mbase;
  const float* st   = stT + mbase;

  __shared__ _Float16 Pbuf[DSZ * LDP];  // ~34 KB, padded rows
  __shared__ float ash[CK];
  if (threadIdx.x < CK) ash[threadIdx.x] = a_cs[bh * S + c * CK + threadIdx.x];

  // warm the K/V tiles (used in later phases) while the inter phase runs
  {
    int rowp = threadIdx.x & 127;  // 256B rows = 2 cache lines each
    __builtin_prefetch(K + (size_t)rowp * DSZ, 0, 0);
    __builtin_prefetch(K + (size_t)rowp * DSZ + 64, 0, 0);
    __builtin_prefetch(V + (size_t)rowp * DSZ, 0, 0);
    __builtin_prefetch(V + (size_t)rowp * DSZ + 64, 0, 0);
  }
  __syncthreads();

  int row0 = wave * 16;
  v16h aq[4];
#pragma unroll
  for (int kt = 0; kt < 4; ++kt) aq[kt] = load_a16(Q, DSZ, row0, kt * 32, lane);

  // ---- inter: acc = Q @ states (states stored p-major) -------------------
  v8f acc[8];
#pragma unroll
  for (int ct = 0; ct < 8; ++ct)
#pragma unroll
    for (int e = 0; e < 8; ++e) acc[ct][e] = 0.f;
#pragma unroll
  for (int kt = 0; kt < 4; ++kt) {
    v16h bops[8];
#pragma unroll
    for (int ct = 0; ct < 8; ++ct) bops[ct] = load_b16_f32(st, DSZ, ct * 16, kt * 32, lane);
#pragma unroll
    for (int ct = 0; ct < 8; ++ct) acc[ct] = wmma_f16(aq[kt], bops[ct], acc[ct]);
  }
  // row scale by exp(a_cs[row])
#pragma unroll
  for (int g = 0; g < 8; ++g) {
    int m = row0 + g + (lane >= 16 ? 8 : 0);
    float e = __expf(ash[m >> 1]);
#pragma unroll
    for (int ct = 0; ct < 8; ++ct) acc[ct][g] *= e;
  }

  // ---- scores P = Q @ K^T, mask+decay, stage f16 in LDS ------------------
#pragma unroll
  for (int ct = 0; ct < 8; ++ct) {
    v16h bops[4];
#pragma unroll
    for (int kt = 0; kt < 4; ++kt) bops[kt] = load_b16(K, DSZ, ct * 16, kt * 32, lane);
    v8f p;
#pragma unroll
    for (int e = 0; e < 8; ++e) p[e] = 0.f;
#pragma unroll
    for (int kt = 0; kt < 4; ++kt) p = wmma_f16(aq[kt], bops[kt], p);

    int nn = ct * 16 + (lane & 15);
    int cj = nn >> 1;
#pragma unroll
    for (int g = 0; g < 8; ++g) {
      int m  = row0 + g + (lane >= 16 ? 8 : 0);
      int ci = m >> 1;
      float w = (ci >= cj) ? __expf(ash[ci] - ash[cj]) : 0.f;
      Pbuf[m * LDP + nn] = (_Float16)(p[g] * w);
    }
  }
  __syncthreads();

  // ---- intra: acc += P @ V  (A from LDS via ds b128) ---------------------
#pragma unroll
  for (int kt = 0; kt < 4; ++kt) {
    v16h ap = load_a16((const _Float16*)Pbuf, LDP, row0, kt * 32, lane);
    v16h bops[8];
#pragma unroll
    for (int ct = 0; ct < 8; ++ct) bops[ct] = load_b16(V, DSZ, ct * 16, kt * 32, lane);
#pragma unroll
    for (int ct = 0; ct < 8; ++ct) acc[ct] = wmma_f16(ap, bops[ct], acc[ct]);
  }

  // ---- fused epilogue -----------------------------------------------------
  float Dh = Dp[h];
#pragma unroll
  for (int ct = 0; ct < 8; ++ct) {
    int nn = ct * 16 + (lane & 15);
#pragma unroll
    for (int gp = 0; gp < 4; ++gp) {  // (r=0,r=1) pairs share a timestep
      int m0 = row0 + 2 * gp + (lane >= 16 ? 8 : 0);
      int t  = c * CK + (m0 >> 1);
      float vraw = v[((size_t)(b * S + t) * HB + h) * DSZ + nn];
      float zraw = z[((size_t)(b * S + t) * HB + h) * DSZ + nn];
      const float* qk4 = qkd + (((size_t)bh) * S + t) * 4;
      float vf0 = vraw * mv[(h * RB + 0) * DSZ + nn];
      float vf1 = vraw * mv[(h * RB + 1) * DSZ + nn];
      float fin = 0.f;
#pragma unroll
      for (int r = 0; r < RB; ++r) {
        float o = acc[ct][2 * gp + r];
        o -= qk4[r * 2 + 0] * vf0 + qk4[r * 2 + 1] * vf1;  // qkv correction
        o += (r == 0 ? vf0 : vf1) * Dh;                    // D * v
        float zc = zraw * mz[(h * RB + r) * DSZ + nn];
        o *= zc / (1.f + __expf(-zc));                     // SiLU gate
        fin += o * mo[(h * RB + r) * DSZ + nn];            // mimo_o reduce
      }
      out[((size_t)(b * S + t) * HB + h) * DSZ + nn] = fin;
    }
  }
}

// ---------------------------------------------------------------------------
extern "C" void kernel_launch(void* const* d_in, const int* in_sizes, int n_in,
                              void* d_out, int out_size, void* d_ws, size_t ws_size,
                              hipStream_t stream) {
  const float* q    = (const float*)d_in[0];
  const float* k    = (const float*)d_in[1];
  const float* v    = (const float*)d_in[2];
  const float* qb   = (const float*)d_in[3];
  const float* kb   = (const float*)d_in[4];
  const float* mv   = (const float*)d_in[5];
  const float* mo   = (const float*)d_in[6];
  const float* z    = (const float*)d_in[7];
  const float* mz   = (const float*)d_in[8];
  const float* ang  = (const float*)d_in[9];
  const float* da   = (const float*)d_in[10];
  const float* dt   = (const float*)d_in[11];
  const float* trap = (const float*)d_in[12];
  const float* Dp   = (const float*)d_in[13];
  float* out = (float*)d_out;

  const int S   = in_sizes[10] / (BB * HB);  // da is (B,H,S)
  const int nch = S / CK;

  // workspace carve-up
  size_t BHS = (size_t)BB * HB * S;
  size_t BHn = (size_t)BB * HB * nch;
  size_t MAT = BHn * (size_t)DSZ * DSZ;
  char* w = (char*)d_ws;
  float* a_cs   = (float*)w;  w += BHS * sizeof(float);
  float* e_rev  = (float*)w;  w += BHS * sizeof(float);
  float* factor = (float*)w;  w += BHS * sizeof(float);
  float* sg     = (float*)w;  w += BHS * sizeof(float);
  float* decay  = (float*)w;  w += BHn * sizeof(float);
  float* qkd    = (float*)w;  w += BHS * 4 * sizeof(float);
  float* kvbuf  = (float*)w;  w += MAT * sizeof(float);
  _Float16* Qh  = (_Float16*)w;  w += MAT * sizeof(_Float16);
  _Float16* Kh  = (_Float16*)w;  w += MAT * sizeof(_Float16);
  _Float16* Krt = (_Float16*)w;  w += MAT * sizeof(_Float16);
  _Float16* Vt  = (_Float16*)w;  w += MAT * sizeof(_Float16);

  k_pre<<<dim3(BB * HB * nch), dim3(CK), 0, stream>>>(da, dt, trap, a_cs, e_rev,
                                                      factor, sg, decay, S, nch);
  k_build<<<dim3(BB * HB * nch), dim3(DSZ), 0, stream>>>(q, k, v, qb, kb, mv, ang,
                                                         factor, sg, e_rev,
                                                         Qh, Kh, Krt, Vt, qkd, S, nch);
  k_kv<<<dim3(BB * HB * nch), dim3(256), 0, stream>>>(Krt, Vt, kvbuf, nch);
  k_scan<<<dim3(BB * HB), dim3(256), 0, stream>>>(kvbuf, decay, nch);
  k_out<<<dim3(BB * HB * nch), dim3(256), 0, stream>>>(Qh, Kh, Vt, kvbuf, a_cs, qkd,
                                                       v, z, mv, mz, mo, Dp, out,
                                                       S, nch);
}
